// EnsembleGATDGFLayer_12549894439562
// MI455X (gfx1250) — compile-verified, matrix operand order checked by reading, not run on hardware
//
#include <hip/hip_runtime.h>
#include <hip/hip_bf16.h>
#include <math.h>

// ---------------------------------------------------------------------------
// Problem constants (match reference)
// ---------------------------------------------------------------------------
#define BB   64
#define NN   512
#define DIN  256
#define DOUT 256
#define DOP  64
#define BN   (BB * NN)          // 32768 rows
#define LN_EPS 1e-5f
#define NEG_SLOPE 0.2f

// ---------------------------------------------------------------------------
// WMMA types (CDNA5 gfx1250, wave32)
// ---------------------------------------------------------------------------
typedef __attribute__((ext_vector_type(16))) __bf16 v16bf;
typedef __attribute__((ext_vector_type(8)))  float  v8f;

union FragBF {
    v16bf        bf;
    unsigned int u32[8];
};

__device__ __forceinline__ unsigned short bf16_bits(float f) {
    unsigned int u = __float_as_uint(f);
    u += 0x7fffu + ((u >> 16) & 1u);   // round-to-nearest-even
    return (unsigned short)(u >> 16);
}

// 16x32 bf16 fragment from a row-major [rows x LD] bf16 buffer.
// ISA 7.12.2 layout: lane r (0..15) holds row M=r, lane-half selects the K
// sub-block, VGPR v holds contiguous K pair -> pure dword loads (merge to b128).
// Used for A fragments (row = m) AND B fragments of NT GEMMs (row = n).
template <int LD>
__device__ __forceinline__ v16bf load_frag(const unsigned short* __restrict__ base,
                                           int row0, int k0, int lane) {
    const int r = lane & 15;
    const int h = (lane >> 4) & 1;
    const unsigned short* p = base + (size_t)(row0 + r) * LD + k0;
    FragBF f;
#pragma unroll
    for (int v = 0; v < 8; ++v) {
        const int kk = ((v < 4) ? 0 : 16) + (h ? 8 : 0) + 2 * (v & 3);
        f.u32[v] = *(const unsigned int*)(p + kk);
    }
    return f.bf;
}

__device__ __forceinline__ v8f wmma_bf16(v16bf a, v16bf b, v8f c) {
    return __builtin_amdgcn_wmma_f32_16x16x32_bf16(
        /*neg_a=*/false, a, /*neg_b=*/false, b,
        /*c_mod=*/(short)0, c, /*reuse_a=*/false, /*reuse_b=*/false);
}

// 32x64 register-blocked NT GEMM core: A [M x K] row-major, B [N x K] row-major.
// 8 WMMA per K-step from 6 fragment loads.
template <int KTOT, int LDA, int LDB>
__device__ __forceinline__ void gemm_32x64(const unsigned short* __restrict__ A, int row0,
                                           const unsigned short* __restrict__ Bn, int n0,
                                           int lane, v8f acc[2][4]) {
#pragma unroll
    for (int k0 = 0; k0 < KTOT; k0 += 32) {
        if (k0 + 32 < KTOT) {
            __builtin_prefetch(A  + (size_t)(row0 + (lane & 15)) * LDA + k0 + 32, 0, 1);
            __builtin_prefetch(Bn + (size_t)(n0   + (lane & 15)) * LDB + k0 + 32, 0, 1);
        }
        v16bf a0 = load_frag<LDA>(A, row0,      k0, lane);
        v16bf a1 = load_frag<LDA>(A, row0 + 16, k0, lane);
        v16bf b0 = load_frag<LDB>(Bn, n0,      k0, lane);
        v16bf b1 = load_frag<LDB>(Bn, n0 + 16, k0, lane);
        v16bf b2 = load_frag<LDB>(Bn, n0 + 32, k0, lane);
        v16bf b3 = load_frag<LDB>(Bn, n0 + 48, k0, lane);
        acc[0][0] = wmma_bf16(a0, b0, acc[0][0]);
        acc[0][1] = wmma_bf16(a0, b1, acc[0][1]);
        acc[0][2] = wmma_bf16(a0, b2, acc[0][2]);
        acc[0][3] = wmma_bf16(a0, b3, acc[0][3]);
        acc[1][0] = wmma_bf16(a1, b0, acc[1][0]);
        acc[1][1] = wmma_bf16(a1, b1, acc[1][1]);
        acc[1][2] = wmma_bf16(a1, b2, acc[1][2]);
        acc[1][3] = wmma_bf16(a1, b3, acc[1][3]);
    }
}

// ---------------------------------------------------------------------------
// K0a: generic f32 -> bf16 convert (inputs, op_emb, adj)
// ---------------------------------------------------------------------------
__global__ void cvt_bf16_kernel(const float* __restrict__ src,
                                unsigned short* __restrict__ dst, size_t n) {
    size_t i = (size_t)blockIdx.x * blockDim.x + threadIdx.x;
    const size_t stride = (size_t)gridDim.x * blockDim.x;
    for (; i < n; i += stride) dst[i] = bf16_bits(src[i]);
}

// ---------------------------------------------------------------------------
// K0b: pack weights to bf16, everything stored N-major [N x K] so all GEMMs
// are NT (contiguous-K fragment loads on both sides).
//   WdT [o][i] = dgf_W[i][o]           (only matrix needing a transpose)
//   WkB/WvB    = Wk/Wv as-is           (already [DOUT x DIN] = [n][k])
//   WqB [o][i] = Wq[o][i] * a_w[o]     (fold attention vector + keep N-major)
//   opWd/opWg  = as-is                 (already [DOUT x DOP] = [n][k])
// ---------------------------------------------------------------------------
__global__ void pack_weights_kernel(const float* __restrict__ dgf_W,
                                    const float* __restrict__ Wk,
                                    const float* __restrict__ Wv,
                                    const float* __restrict__ Wq,
                                    const float* __restrict__ a_w,
                                    const float* __restrict__ dgf_opW,
                                    const float* __restrict__ gat_opW,
                                    unsigned short* __restrict__ WdT,
                                    unsigned short* __restrict__ WkB,
                                    unsigned short* __restrict__ WvB,
                                    unsigned short* __restrict__ WqB,
                                    unsigned short* __restrict__ opWd,
                                    unsigned short* __restrict__ opWg) {
    const int t = blockIdx.x * blockDim.x + threadIdx.x;   // 0 .. 65535
    const int o = t >> 8;      // output-feature (N) index
    const int i = t & 255;     // input-feature  (K) index
    WdT[t] = bf16_bits(dgf_W[(size_t)i * DOUT + o]);
    WkB[t] = bf16_bits(Wk[t]);
    WvB[t] = bf16_bits(Wv[t]);
    WqB[t] = bf16_bits(Wq[t] * a_w[o]);
    if (i < DOP) {
        opWd[o * DOP + i] = bf16_bits(dgf_opW[(size_t)o * DOP + i]);
        opWg[o * DOP + i] = bf16_bits(gat_opW[(size_t)o * DOP + i]);
    }
}

// ---------------------------------------------------------------------------
// K1: four projections.  grid = (BN/32, DOUT/64, 4), block = 32.
//   mat 0: support = X @ dgf_W   -> SUPf (f32 row-major) + SUPt (bf16 [B][O][N])
//   mat 1: Whk                   -> WHK  (bf16 row-major, A/B of NT score GEMM)
//   mat 2: Whv                   -> WHVt (bf16 [B][O][N] for NT aggregation)
//   mat 3: Whq*a_w               -> WHQ  (bf16 row-major)
// ---------------------------------------------------------------------------
__global__ void proj_kernel(const unsigned short* __restrict__ Xbf,
                            const unsigned short* __restrict__ WdT,
                            const unsigned short* __restrict__ WkB,
                            const unsigned short* __restrict__ WvB,
                            const unsigned short* __restrict__ WqB,
                            float* __restrict__ SUPf,
                            unsigned short* __restrict__ SUPt,
                            unsigned short* __restrict__ WHK,
                            unsigned short* __restrict__ WHVt,
                            unsigned short* __restrict__ WHQ) {
    const int lane = threadIdx.x;
    const int row0 = blockIdx.x * 32;
    const int n0   = blockIdx.y * 64;
    const int mat  = blockIdx.z;
    const unsigned short* W = (mat == 0) ? WdT : (mat == 1) ? WkB
                                               : (mat == 2) ? WvB : WqB;
    v8f acc[2][4] = {};
    gemm_32x64<DIN, DIN, DIN>(Xbf, row0, W, n0, lane, acc);

    const int r = lane & 15;
    const int h = (lane >> 4) & 1;
#pragma unroll
    for (int i2 = 0; i2 < 2; ++i2)
#pragma unroll
        for (int j4 = 0; j4 < 4; ++j4)
#pragma unroll
            for (int i = 0; i < 8; ++i) {
                const int m = row0 + 16 * i2 + i + 8 * h;   // global row in [0,BN)
                const int n = n0 + 16 * j4 + r;
                const float v = acc[i2][j4][i];
                const size_t rowmaj = (size_t)m * DOUT + n;
                const size_t colmaj =
                    ((size_t)(m >> 9) * DOUT + n) * NN + (m & (NN - 1));
                if (mat == 0) { SUPf[rowmaj] = v; SUPt[colmaj] = bf16_bits(v); }
                else if (mat == 1) WHK[rowmaj]  = bf16_bits(v);
                else if (mat == 2) WHVt[colmaj] = bf16_bits(v);
                else               WHQ[rowmaj]  = bf16_bits(v);
            }
}

// ---------------------------------------------------------------------------
// K1b: gate GEMMs (K=64) + fused sigmoid.  grid = (BN/16, DOUT/64, 2).
// 16x64 tile: A fragment reused across 4 B fragments.
// ---------------------------------------------------------------------------
__global__ void gates_kernel(const unsigned short* __restrict__ Ebf,
                             const unsigned short* __restrict__ opWd,
                             const unsigned short* __restrict__ opWg,
                             const float* __restrict__ opbd,
                             const float* __restrict__ opbg,
                             float* __restrict__ GD,
                             float* __restrict__ GG) {
    const int lane = threadIdx.x;
    const int row0 = blockIdx.x * 16;
    const int n0   = blockIdx.y * 64;
    const int mat  = blockIdx.z;
    const unsigned short* W = mat ? opWg : opWd;
    const float* bias       = mat ? opbg : opbd;
    float* G                = mat ? GG   : GD;
    v8f acc[4] = {};
#pragma unroll
    for (int k0 = 0; k0 < DOP; k0 += 32) {
        v16bf a  = load_frag<DOP>(Ebf, row0, k0, lane);
#pragma unroll
        for (int j4 = 0; j4 < 4; ++j4) {
            v16bf b = load_frag<DOP>(W, n0 + 16 * j4, k0, lane);
            acc[j4] = wmma_bf16(a, b, acc[j4]);
        }
    }
    const int r = lane & 15;
    const int h = (lane >> 4) & 1;
#pragma unroll
    for (int j4 = 0; j4 < 4; ++j4)
#pragma unroll
        for (int i = 0; i < 8; ++i) {
            const int n = n0 + 16 * j4 + r;
            const size_t idx = (size_t)(row0 + i + 8 * h) * DOUT + n;
            const float x = acc[j4][i] + bias[n];
            G[idx] = 1.0f / (1.0f + __expf(-x));
        }
}

// ---------------------------------------------------------------------------
// K2: attention scores per batch:  S[b,e,l] = sum_m WHQ[b,e,m] * WHK[b,l,m].
// NT GEMM, 32x64 tile.  grid = (N/32, N/64, B), block = 32.
// ---------------------------------------------------------------------------
__global__ void scores_kernel(const unsigned short* __restrict__ WHQ,
                              const unsigned short* __restrict__ WHK,
                              float* __restrict__ SC) {
    const int lane = threadIdx.x;
    const int e0 = blockIdx.x * 32;
    const int l0 = blockIdx.y * 64;
    const int b  = blockIdx.z;
    const unsigned short* Q = WHQ + (size_t)b * NN * DOUT;
    const unsigned short* K = WHK + (size_t)b * NN * DOUT;
    v8f acc[2][4] = {};
    gemm_32x64<DOUT, DOUT, DOUT>(Q, e0, K, l0, lane, acc);

    const int r = lane & 15;
    const int h = (lane >> 4) & 1;
    float* out = SC + (size_t)b * NN * NN;
#pragma unroll
    for (int i2 = 0; i2 < 2; ++i2)
#pragma unroll
        for (int j4 = 0; j4 < 4; ++j4)
#pragma unroll
            for (int i = 0; i < 8; ++i)
                out[(size_t)(e0 + 16 * i2 + i + 8 * h) * NN + l0 + 16 * j4 + r] =
                    acc[i2][j4][i];
}

// ---------------------------------------------------------------------------
// K3: per-row  s/16 -> leaky_relu -> * adj -> softmax -> bf16 attn.
// One wave per row (512 cols = 16 per lane), shuffle reductions.
// ---------------------------------------------------------------------------
__global__ void softmax_kernel(const float* __restrict__ SC,
                               const float* __restrict__ adj,
                               unsigned short* __restrict__ ATTN) {
    const int tid  = threadIdx.x;
    const int lane = tid & 31;
    const int w    = tid >> 5;
    const size_t row = (size_t)blockIdx.x * 8 + w;     // 0..BN-1
    const float* s  = SC  + row * NN;
    const float* am = adj + row * NN;
    float vals[16];
    float mx = -3.4e38f;
#pragma unroll
    for (int j = 0; j < 16; ++j) {
        float x = s[j * 32 + lane] * 0.0625f;          // 1/sqrt(DOUT)
        x = (x > 0.0f) ? x : NEG_SLOPE * x;            // leaky_relu
        x *= am[j * 32 + lane];                        // multiplicative mask
        vals[j] = x;
        mx = fmaxf(mx, x);
    }
#pragma unroll
    for (int off = 16; off > 0; off >>= 1) mx = fmaxf(mx, __shfl_xor(mx, off, 32));
    float sum = 0.0f;
#pragma unroll
    for (int j = 0; j < 16; ++j) { vals[j] = __expf(vals[j] - mx); sum += vals[j]; }
#pragma unroll
    for (int off = 16; off > 0; off >>= 1) sum += __shfl_xor(sum, off, 32);
    const float inv = 1.0f / sum;
    unsigned short* o = ATTN + row * NN;
#pragma unroll
    for (int j = 0; j < 16; ++j) o[j * 32 + lane] = bf16_bits(vals[j] * inv);
}

// ---------------------------------------------------------------------------
// K4: neighborhood aggregations (NT GEMM, K = 512, 32x64 tile).
//   mat 0: AS[b,i,o] = sum_j adj_bf[b,i,j]  * SUPt[b,o,j]
//   mat 1: HV[b,e,m] = sum_l attn_bf[b,e,l] * WHVt[b,m,l]
// grid = (N/32, DOUT/64, 2*B), block = 32.
// ---------------------------------------------------------------------------
__global__ void agg_kernel(const unsigned short* __restrict__ ADJbf,
                           const unsigned short* __restrict__ ATTN,
                           const unsigned short* __restrict__ SUPt,
                           const unsigned short* __restrict__ WHVt,
                           float* __restrict__ AS,
                           float* __restrict__ HV) {
    const int lane = threadIdx.x;
    const int r0 = blockIdx.x * 32;
    const int n0 = blockIdx.y * 64;
    const int z  = blockIdx.z;
    const int mat = z >> 6;
    const int b   = z & 63;
    const unsigned short* A  = (mat ? ATTN : ADJbf) + (size_t)b * NN * NN;
    const unsigned short* Bn = (mat ? WHVt : SUPt)  + (size_t)b * DOUT * NN;
    float* O = mat ? HV : AS;
    v8f acc[2][4] = {};
    gemm_32x64<NN, NN, NN>(A, r0, Bn, n0, lane, acc);

    const int r = lane & 15;
    const int h = (lane >> 4) & 1;
#pragma unroll
    for (int i2 = 0; i2 < 2; ++i2)
#pragma unroll
        for (int j4 = 0; j4 < 4; ++j4)
#pragma unroll
            for (int i = 0; i < 8; ++i)
                O[((size_t)b * NN + r0 + 16 * i2 + i + 8 * h) * DOUT +
                  n0 + 16 * j4 + r] = acc[i2][j4][i];
}

// ---------------------------------------------------------------------------
// K5: fused epilogue. One wave per row; LayerNorm via shuffle reductions.
// ---------------------------------------------------------------------------
__global__ void final_kernel(const float* __restrict__ SUPf,
                             const float* __restrict__ AS,
                             const float* __restrict__ HV,
                             const float* __restrict__ GD,
                             const float* __restrict__ GG,
                             const float* __restrict__ dgf_b,
                             const float* __restrict__ ln_g,
                             const float* __restrict__ ln_b,
                             float* __restrict__ out) {
    const int tid  = threadIdx.x;
    const int lane = tid & 31;
    const int w    = tid >> 5;
    const size_t row  = (size_t)blockIdx.x * 8 + w;
    const size_t base = row * DOUT;
    float dn[8], hp[8];
    float s1 = 0.0f, s2 = 0.0f;
#pragma unroll
    for (int j = 0; j < 8; ++j) {
        const int c = j * 32 + lane;
        const float sup = SUPf[base + c];
        dn[j] = GD[base + c] * AS[base + c] + sup + dgf_b[c];
        const float hh = GG[base + c] * HV[base + c];
        hp[j] = hh;
        s1 += hh;
        s2 += hh * hh;
    }
#pragma unroll
    for (int off = 16; off > 0; off >>= 1) {
        s1 += __shfl_xor(s1, off, 32);
        s2 += __shfl_xor(s2, off, 32);
    }
    const float mu  = s1 * (1.0f / DOUT);
    const float var = s2 * (1.0f / DOUT) - mu * mu;
    const float rs  = rsqrtf(var + LN_EPS);
#pragma unroll
    for (int j = 0; j < 8; ++j) {
        const int c = j * 32 + lane;
        const float g = (hp[j] - mu) * rs * ln_g[c] + ln_b[c];
        out[base + c] = 0.5f * (dn[j] + g);
    }
}

// ---------------------------------------------------------------------------
// Host launcher
// ---------------------------------------------------------------------------
extern "C" void kernel_launch(void* const* d_in, const int* in_sizes, int n_in,
                              void* d_out, int out_size, void* d_ws, size_t ws_size,
                              hipStream_t stream) {
    (void)in_sizes; (void)n_in; (void)out_size; (void)ws_size;

    const float* inputs  = (const float*)d_in[0];
    const float* adj     = (const float*)d_in[1];
    const float* op_emb  = (const float*)d_in[2];
    const float* dgf_W   = (const float*)d_in[3];
    const float* dgf_b   = (const float*)d_in[4];
    const float* dgf_opW = (const float*)d_in[5];
    const float* dgf_opb = (const float*)d_in[6];
    const float* Wk      = (const float*)d_in[7];
    const float* Wv      = (const float*)d_in[8];
    const float* Wq      = (const float*)d_in[9];
    const float* a_w     = (const float*)d_in[10];
    const float* gat_opW = (const float*)d_in[11];
    const float* gat_opb = (const float*)d_in[12];
    const float* ln_g    = (const float*)d_in[13];
    const float* ln_b    = (const float*)d_in[14];
    float* out = (float*)d_out;

    // -------- workspace carve-up --------
    char* ws = (char*)d_ws;
    size_t off = 0;
    auto alloc = [&](size_t bytes) {
        void* p = ws + off;
        off += (bytes + 255) & ~(size_t)255;
        return p;
    };
    unsigned short* Xbf   = (unsigned short*)alloc((size_t)BN * DIN * 2);
    unsigned short* Ebf   = (unsigned short*)alloc((size_t)BN * DOP * 2);
    unsigned short* ADJbf = (unsigned short*)alloc((size_t)BB * NN * NN * 2);
    unsigned short* WdT   = (unsigned short*)alloc((size_t)DIN * DOUT * 2);
    unsigned short* WkB   = (unsigned short*)alloc((size_t)DIN * DOUT * 2);
    unsigned short* WvB   = (unsigned short*)alloc((size_t)DIN * DOUT * 2);
    unsigned short* WqB   = (unsigned short*)alloc((size_t)DIN * DOUT * 2);
    unsigned short* opWd  = (unsigned short*)alloc((size_t)DOP * DOUT * 2);
    unsigned short* opWg  = (unsigned short*)alloc((size_t)DOP * DOUT * 2);
    float*          SUPf  = (float*)alloc((size_t)BN * DOUT * 4);
    unsigned short* SUPt  = (unsigned short*)alloc((size_t)BN * DOUT * 2);
    unsigned short* WHK   = (unsigned short*)alloc((size_t)BN * DOUT * 2);
    unsigned short* WHVt  = (unsigned short*)alloc((size_t)BN * DOUT * 2);
    unsigned short* WHQ   = (unsigned short*)alloc((size_t)BN * DOUT * 2);
    float*          GD    = (float*)alloc((size_t)BN * DOUT * 4);
    float*          GG    = (float*)alloc((size_t)BN * DOUT * 4);
    float*          SC    = (float*)alloc((size_t)BB * NN * NN * 4);
    unsigned short* ATTN  = (unsigned short*)alloc((size_t)BB * NN * NN * 2);
    // AS/HV alias the scores buffer (scores fully consumed by softmax first).
    float* AS = (float*)SC;
    float* HV = (float*)(SC + (size_t)BN * DOUT);

    // -------- stage 0: precision conversion & weight packing --------
    cvt_bf16_kernel<<<4096, 256, 0, stream>>>(inputs, Xbf, (size_t)BN * DIN);
    cvt_bf16_kernel<<<2048, 256, 0, stream>>>(op_emb, Ebf, (size_t)BN * DOP);
    cvt_bf16_kernel<<<8192, 256, 0, stream>>>(adj, ADJbf, (size_t)BB * NN * NN);
    pack_weights_kernel<<<(DIN * DOUT) / 256, 256, 0, stream>>>(
        dgf_W, Wk, Wv, Wq, a_w, dgf_opW, gat_opW,
        WdT, WkB, WvB, WqB, opWd, opWg);

    // -------- stage 1: projections + gates (WMMA) --------
    proj_kernel<<<dim3(BN / 32, DOUT / 64, 4), 32, 0, stream>>>(
        Xbf, WdT, WkB, WvB, WqB, SUPf, SUPt, WHK, WHVt, WHQ);
    gates_kernel<<<dim3(BN / 16, DOUT / 64, 2), 32, 0, stream>>>(
        Ebf, opWd, opWg, dgf_opb, gat_opb, GD, GG);

    // -------- stage 2: attention scores (WMMA) + softmax --------
    scores_kernel<<<dim3(NN / 32, NN / 64, BB), 32, 0, stream>>>(WHQ, WHK, SC);
    softmax_kernel<<<BN / 8, 256, 0, stream>>>(SC, adj, ATTN);

    // -------- stage 3: neighborhood aggregations (WMMA) --------
    agg_kernel<<<dim3(NN / 32, DOUT / 64, 2 * BB), 32, 0, stream>>>(
        ADJbf, ATTN, SUPt, WHVt, AS, HV);

    // -------- stage 4: fused gating + residual + LayerNorm + combine --------
    final_kernel<<<BN / 8, 256, 0, stream>>>(
        SUPf, AS, HV, GD, GG, dgf_b, ln_g, ln_b, out);
}
